// ConduitDecoder_30090540875901
// MI455X (gfx1250) — matrix operand order repaired and checked by reading, not compiled, not run on hardware
//
#include <hip/hip_runtime.h>
#include <math.h>

typedef _Float16 half_t;
typedef __attribute__((ext_vector_type(16))) _Float16 v16h;
typedef __attribute__((ext_vector_type(8)))  float    v8f;
typedef __attribute__((ext_vector_type(8)))  unsigned u32x8;
typedef __attribute__((ext_vector_type(4)))  unsigned u32x4;
typedef __attribute__((ext_vector_type(8)))  int      i32x8;
typedef __attribute__((ext_vector_type(4)))  int      i32x4;

// ---------------- constants ----------------
#define BN    8192
#define VDIM  64
#define DDIM  16
#define HDIM  512
#define DEPTH 4
#define PDIM  768
#define MD    85
#define KPAD  96
#define MROWS (BN * DDIM)   // 131072
#define DH    (DDIM * HDIM) // 8192

// epilogue modes
#define EPI_GELU16 0   // gelu -> f16
#define EPI_RES    1   // h += acc + b   -> f32 + f16 copy
#define EPI_OUT    2   // acc + b        -> f32 (d_out)
#define EPI_LN16   3   // gelu -> LN     -> f16
#define EPI_FUSE   4   // rank2 + gelu -> LN -> f32

__device__ __forceinline__ float gelu_exact(float x) {
    return 0.5f * x * (1.0f + erff(x * 0.70710678118654752f));
}

// -------- fragment loaders (documented CDNA5 16-bit layouts, b128 LDS loads) ---
// A 16x32 (MxK): lane l holds row M=l&15; hi=l>>4 selects K halves 8..15/24..31.
// Row is 64B; needed dwords are [hi*4..hi*4+3] and [8+hi*4..8+hi*4+3].
__device__ __forceinline__ v16h load_a_frag(const half_t* sA, int lane) {
    const int m = lane & 15, hi = lane >> 4;
    const u32x4* base = (const u32x4*)(sA + m * 32);   // 64B-aligned rows
    const u32x4 lo = base[hi];
    const u32x4 up = base[2 + hi];
    u32x8 r;
    r[0] = lo[0]; r[1] = lo[1]; r[2] = lo[2]; r[3] = lo[3];
    r[4] = up[0]; r[5] = up[1]; r[6] = up[2]; r[7] = up[3];
    return __builtin_bit_cast(v16h, r);
}
// B 32x16 (KxN): lane l holds col N=l&15; hi=l>>4 selects K halves 8..15/24..31.
// sB packed per output column: sB[n*32 + k]; needed dwords are [hi*4..hi*4+3] and [8+hi*4..8+hi*4+3].
__device__ __forceinline__ v16h load_b_frag(const half_t* sB, int n0, int lane) {
    const int n = n0 + (lane & 15), hi = lane >> 4;
    const u32x4* base = (const u32x4*)(sB + n * 32);   // 64B-aligned rows
    const u32x4 lo = base[hi];
    const u32x4 up = base[2 + hi];
    u32x8 r;
    r[0] = lo[0]; r[1] = lo[1]; r[2] = lo[2]; r[3] = lo[3];
    r[4] = up[0]; r[5] = up[1]; r[6] = up[2]; r[7] = up[3];
    return __builtin_bit_cast(v16h, r);
}

// -------- Tensor Data Mover: 2D f16 tile (rows x 32 elems) -> LDS --------
#define HAVE_TDM __has_builtin(__builtin_amdgcn_tensor_load_to_lds)

__device__ __forceinline__ void tdm_load_2d(const void* gaddr, unsigned lds_off,
                                            unsigned rowElems, unsigned rows,
                                            unsigned strideElems,
                                            unsigned tensor0, unsigned tensor1) {
#if HAVE_TDM
    const unsigned long long ga = (unsigned long long)gaddr;
    u32x4 g0;
    g0[0] = 1u;                                                // count=1, user desc
    g0[1] = lds_off;                                           // LDS byte address
    g0[2] = (unsigned)(ga & 0xffffffffu);                      // global addr [31:0]
    g0[3] = (unsigned)((ga >> 32) & 0x01ffffffu) | (2u << 30); // addr[56:32] | type=2
    i32x8 g1;
    g1[0] = 0x00010000;                                        // data_size=2B, mask=0
    g1[1] = (int)((tensor0 & 0xffffu) << 16);                  // tensor_dim0[15:0]
    g1[2] = (int)((tensor0 >> 16) | ((tensor1 & 0xffffu) << 16));
    g1[3] = (int)((tensor1 >> 16) | (rowElems << 16));         // t1[31:16]|tile_dim0
    g1[4] = (int)(rows & 0xffffu);                             // tile_dim1 (tile_dim2=0)
    g1[5] = (int)strideElems;                                  // tensor_dim0_stride lo
    g1[6] = 0;
    g1[7] = 0;
    i32x4 z4; z4[0] = 0; z4[1] = 0; z4[2] = 0; z4[3] = 0;
#if defined(__clang_major__) && (__clang_major__ >= 23)
    i32x8 z8;
#pragma unroll
    for (int i = 0; i < 8; ++i) z8[i] = 0;
    __builtin_amdgcn_tensor_load_to_lds(g0, g1, z4, z4, z8, 0);
#else
    __builtin_amdgcn_tensor_load_to_lds(g0, g1, z4, z4, 0);
#endif
#else
    (void)gaddr; (void)lds_off; (void)rowElems; (void)rows;
    (void)strideElems; (void)tensor0; (void)tensor1;
#endif
}

// ---------------- prep kernels ----------------
__global__ void cvt_pad_kernel(half_t* __restrict__ dst, const float* __restrict__ src,
                               int rows, int dstK, int srcK) {
    size_t idx = (size_t)blockIdx.x * 256 + threadIdx.x;
    if (idx >= (size_t)rows * dstK) return;
    int n = (int)(idx / dstK), k = (int)(idx % dstK);
    float v = (k < srcK) ? src[(size_t)n * srcK + k] : 0.0f;
    dst[idx] = (half_t)v;
}

__global__ void build_bundle_kernel(half_t* __restrict__ bundle,
                                    const float* __restrict__ U,  const float* __restrict__ S,
                                    const float* __restrict__ Vt, const float* __restrict__ fr,
                                    const float* __restrict__ st, const float* __restrict__ cc,
                                    const float* __restrict__ eo) {
    size_t idx = (size_t)blockIdx.x * 256 + threadIdx.x;
    if (idx >= (size_t)MROWS * KPAD) return;
    int m = (int)(idx / KPAD), k = (int)(idx % KPAD);
    int b = m >> 4, d = m & 15, bd = b * DDIM + d;
    float v;
    if      (k < VDIM)     v = U[((size_t)b * VDIM + k) * DDIM + d];
    else if (k == VDIM)    v = S[bd];
    else if (k < 81)       v = Vt[(size_t)b * (DDIM * DDIM) + d * DDIM + (k - 65)];
    else if (k == 81)      v = log1pf(fr[bd]);
    else if (k == 82)      v = st[bd];
    else if (k == 83)      v = cc[bd];
    else if (k == 84)      v = eo[bd];
    else                   v = 0.0f;
    bundle[idx] = (half_t)v;
}

// ---------------- standalone LN over 512 (h f32 -> u f16) ----------------
__global__ void __launch_bounds__(256) ln512_kernel(const float* __restrict__ h,
                                                    const float* __restrict__ g,
                                                    const float* __restrict__ b,
                                                    half_t* __restrict__ u) {
    const int row = blockIdx.x, tid = threadIdx.x;
    const float* hr = h + (size_t)row * HDIM;
    float x0 = hr[tid], x1 = hr[tid + 256];
    float s = x0 + x1, q = x0 * x0 + x1 * x1;
#pragma unroll
    for (int o = 16; o; o >>= 1) { s += __shfl_xor(s, o, 32); q += __shfl_xor(q, o, 32); }
    __shared__ float ss[8], sq[8], smv[2];
    const int wave = tid >> 5, lane = tid & 31;
    if (lane == 0) { ss[wave] = s; sq[wave] = q; }
    __syncthreads();
    if (tid == 0) {
        float S = 0, Q = 0;
        for (int i = 0; i < 8; ++i) { S += ss[i]; Q += sq[i]; }
        float m = S * (1.0f / HDIM);
        smv[0] = m; smv[1] = rsqrtf(Q * (1.0f / HDIM) - m * m + 1e-5f);
    }
    __syncthreads();
    float m = smv[0], rs = smv[1];
    half_t* ur = u + (size_t)row * HDIM;
    ur[tid]       = (half_t)((x0 - m) * rs * g[tid]       + b[tid]);
    ur[tid + 256] = (half_t)((x1 - m) * rs * g[tid + 256] + b[tid + 256]);
}

// ------- generic WMMA GEMM: ROWS x N per workgroup, TDM double-buffered -------
// C[m][n] = sum_k A[m][k] * W[n][k],  A f16 (lda), W f16 row-major (N x K).
template <int N, int ROWS, int EPI>
__global__ void __launch_bounds__(256, 1) gemm_kernel(
    const half_t* __restrict__ A, int lda,
    const half_t* __restrict__ Bw, int K,
    const float* __restrict__ bias,
    const float* __restrict__ g, const float* __restrict__ beta,
    const float* __restrict__ r1, const float* __restrict__ r2,
    const float* __restrict__ WfRaw,   // fuse rank-2 columns (f32, stride 8194)
    const float* hin, float* out32, half_t* out16) {

    constexpr int TPW  = N / 128;            // 16x16 col-tiles per wave
    constexpr int RT   = ROWS / 16;          // row-tiles per workgroup
    constexpr int BUFB = ROWS * 64 + N * 64; // bytes per (A,B) LDS buffer

    extern __shared__ char smem[];           // 2*BUFB bytes, sized at launch

    const int tid = threadIdx.x, wave = tid >> 5, lane = tid & 31;
    const int hi = lane >> 4, l16 = lane & 15;
    const int rowBase = blockIdx.x * ROWS;

    v8f acc[RT][TPW];
#pragma unroll
    for (int rt = 0; rt < RT; ++rt)
#pragma unroll
        for (int t = 0; t < TPW; ++t)
            acc[rt][t] = v8f{0.f, 0.f, 0.f, 0.f, 0.f, 0.f, 0.f, 0.f};

    const int nk = K >> 5;

#if HAVE_TDM
    const unsigned ldsBase = (unsigned)(unsigned long long)(void*)smem;
    if (wave == 0) {   // prologue: stage k-step 0 into buffer 0
        tdm_load_2d(A + (size_t)rowBase * lda, ldsBase,
                    32u, (unsigned)ROWS, (unsigned)lda, (unsigned)lda, (unsigned)ROWS);
        tdm_load_2d(Bw, ldsBase + ROWS * 64,
                    32u, (unsigned)N, (unsigned)K, (unsigned)K, (unsigned)N);
    }
#endif

    for (int kt = 0; kt < nk; ++kt) {
        const int cur = kt & 1;
        const half_t* sA = (const half_t*)(smem + cur * BUFB);
        const half_t* sB = (const half_t*)(smem + cur * BUFB + ROWS * 64);
#if HAVE_TDM
        if (wave == 0) {
            if (kt + 1 < nk) {   // issue next k-step into the other buffer
                const unsigned nb = ldsBase + ((kt + 1) & 1) * BUFB;
                tdm_load_2d(A + (size_t)rowBase * lda + (kt + 1) * 32, nb,
                            32u, (unsigned)ROWS, (unsigned)lda, (unsigned)lda, (unsigned)ROWS);
                tdm_load_2d(Bw + (size_t)(kt + 1) * 32, nb + ROWS * 64,
                            32u, (unsigned)N, (unsigned)K, (unsigned)K, (unsigned)N);
                __builtin_amdgcn_s_wait_tensorcnt(2);  // current pair retired (in-order)
            } else {
                __builtin_amdgcn_s_wait_tensorcnt(0);
            }
        }
        __syncthreads();   // current buffer visible to all waves
#else
        __syncthreads();
        for (int idx = tid; idx < ROWS * 16; idx += 256) {
            const int r = idx >> 4, c2 = idx & 15;
            ((unsigned*)sA)[r * 16 + c2] =
                *(const unsigned*)(A + (size_t)(rowBase + r) * lda + kt * 32 + c2 * 2);
        }
        for (int idx = tid; idx < N * 16; idx += 256) {
            const int n = idx >> 4, c2 = idx & 15;
            ((unsigned*)sB)[n * 16 + c2] =
                *(const unsigned*)(Bw + (size_t)n * K + kt * 32 + c2 * 2);
        }
        __syncthreads();
#endif
        v16h a[RT];
#pragma unroll
        for (int rt = 0; rt < RT; ++rt) a[rt] = load_a_frag(sA + rt * 16 * 32, lane);
#pragma unroll
        for (int t = 0; t < TPW; ++t) {
            const v16h b = load_b_frag(sB, (wave * TPW + t) * 16, lane);
#pragma unroll
            for (int rt = 0; rt < RT; ++rt)
                acc[rt][t] = __builtin_amdgcn_wmma_f32_16x16x32_f16(
                    false, a[rt], false, b, (short)0, acc[rt][t], false, false);
        }
        __syncthreads();   // all reads of this buffer done before it is re-targeted
    }

    if constexpr (EPI == EPI_GELU16) {
#pragma unroll
        for (int rt = 0; rt < RT; ++rt)
#pragma unroll
            for (int t = 0; t < TPW; ++t) {
                const int n = (wave * TPW + t) * 16 + l16;
#pragma unroll
                for (int j = 0; j < 8; ++j) {
                    const int m = rowBase + rt * 16 + j + (hi << 3);
                    out16[(size_t)m * N + n] = (half_t)gelu_exact(acc[rt][t][j] + bias[n]);
                }
            }
    } else if constexpr (EPI == EPI_RES) {
#pragma unroll
        for (int rt = 0; rt < RT; ++rt)
#pragma unroll
            for (int t = 0; t < TPW; ++t) {
                const int n = (wave * TPW + t) * 16 + l16;
#pragma unroll
                for (int j = 0; j < 8; ++j) {
                    const size_t o = (size_t)(rowBase + rt * 16 + j + (hi << 3)) * N + n;
                    const float v = hin[o] + acc[rt][t][j] + bias[n];
                    out32[o] = v;
                    out16[o] = (half_t)v;
                }
            }
    } else if constexpr (EPI == EPI_OUT) {
#pragma unroll
        for (int rt = 0; rt < RT; ++rt)
#pragma unroll
            for (int t = 0; t < TPW; ++t) {
                const int n = (wave * TPW + t) * 16 + l16;
#pragma unroll
                for (int j = 0; j < 8; ++j)
                    out32[(size_t)(rowBase + rt * 16 + j + (hi << 3)) * N + n] =
                        acc[rt][t][j] + bias[n];
            }
    } else {  // EPI_LN16 / EPI_FUSE (N == 512): 16-row LN passes through LDS
        float* sC  = (float*)smem;              // 16 x 512 f32 (overlays buffer 0)
        float* sMV = (float*)(smem + 32768);
        for (int rt = 0; rt < RT; ++rt) {
            __syncthreads();
#pragma unroll
            for (int t = 0; t < TPW; ++t) {
                const int n = (wave * TPW + t) * 16 + l16;
#pragma unroll
                for (int j = 0; j < 8; ++j) {
                    const int mloc = j + (hi << 3);
                    float z = acc[rt][t][j] + bias[n];
                    if constexpr (EPI == EPI_FUSE) {
                        const int m = rowBase + rt * 16 + mloc;
                        z += r1[m] * WfRaw[(size_t)n * 8194 + 8192]
                           + r2[m] * WfRaw[(size_t)n * 8194 + 8193];
                    }
                    sC[mloc * 512 + n] = gelu_exact(z);
                }
            }
            __syncthreads();
            const int row = tid >> 4, sub = tid & 15;
            float s = 0.0f, q = 0.0f;
            for (int c = sub; c < 512; c += 16) {
                const float x = sC[row * 512 + c];
                s += x; q += x * x;
            }
#pragma unroll
            for (int o = 8; o; o >>= 1) { s += __shfl_xor(s, o, 16); q += __shfl_xor(q, o, 16); }
            if (sub == 0) {
                const float mean = s * (1.0f / 512.0f);
                sMV[row * 2]     = mean;
                sMV[row * 2 + 1] = rsqrtf(q * (1.0f / 512.0f) - mean * mean + 1e-5f);
            }
            __syncthreads();
            const float mean = sMV[row * 2], rs = sMV[row * 2 + 1];
            for (int c = sub; c < 512; c += 16) {
                const float y = (sC[row * 512 + c] - mean) * rs * g[c] + beta[c];
                const size_t o = (size_t)(rowBase + rt * 16 + row) * 512 + c;
                if constexpr (EPI == EPI_FUSE) out32[o] = y;
                else                           out16[o] = (half_t)y;
            }
        }
    }
}

// ---------------- host launch ----------------
extern "C" void kernel_launch(void* const* d_in, const int* in_sizes, int n_in,
                              void* d_out, int out_size, void* d_ws, size_t ws_size,
                              hipStream_t stream) {
    (void)in_sizes; (void)n_in; (void)out_size; (void)ws_size;
    const float* U        = (const float*)d_in[0];
    const float* S        = (const float*)d_in[1];
    const float* Vt       = (const float*)d_in[2];
    const float* friction = (const float*)d_in[3];
    const float* settle   = (const float*)d_in[4];
    const float* charc    = (const float*)d_in[5];
    const float* extord   = (const float*)d_in[6];
    const float* refine   = (const float*)d_in[7];
    const float* release  = (const float*)d_in[8];
    const float* Wp   = (const float*)d_in[9];
    const float* bp   = (const float*)d_in[10];
    const float* gp   = (const float*)d_in[11];
    const float* betap = (const float*)d_in[12];
    const float* Wf   = (const float*)d_in[13];
    const float* bf   = (const float*)d_in[14];
    const float* gf   = (const float*)d_in[15];
    const float* betaf = (const float*)d_in[16];
    const float* ln_g = (const float*)d_in[17];
    const float* ln_b = (const float*)d_in[18];
    const float* W1   = (const float*)d_in[19];
    const float* b1   = (const float*)d_in[20];
    const float* W2   = (const float*)d_in[21];
    const float* b2   = (const float*)d_in[22];
    const float* Wo   = (const float*)d_in[23];
    const float* bo   = (const float*)d_in[24];
    float* out = (float*)d_out;

    char* ws = (char*)d_ws;
    size_t off = 0;
    auto carve = [&](size_t bytes) { char* p = ws + off; off = (off + bytes + 255) & ~(size_t)255; return p; };
    half_t* bundle16 = (half_t*)carve((size_t)MROWS * KPAD * 2);
    half_t* fused16  = (half_t*)carve((size_t)MROWS * HDIM * 2);
    float*  h32      = (float*) carve((size_t)BN * HDIM * 4);
    half_t* h16      = (half_t*)carve((size_t)BN * HDIM * 2);
    half_t* u16      = (half_t*)carve((size_t)BN * HDIM * 2);
    half_t* t16      = (half_t*)carve((size_t)BN * HDIM * 2);
    half_t* Wp16     = (half_t*)carve((size_t)HDIM * KPAD * 2);
    half_t* Wf16     = (half_t*)carve((size_t)HDIM * DH * 2);
    half_t* W1_16    = (half_t*)carve((size_t)DEPTH * HDIM * HDIM * 2);
    half_t* W2_16    = (half_t*)carve((size_t)DEPTH * HDIM * HDIM * 2);
    half_t* Wo16     = (half_t*)carve((size_t)PDIM * HDIM * 2);

    auto blocks = [](size_t n) { return (unsigned)((n + 255) / 256); };

    // dynamic LDS: two (A,B) buffers
    const unsigned SM_512_64 = 2u * (64 * 64 + 512 * 64);  // 73728 B
    const unsigned SM_768_32 = 2u * (32 * 64 + 768 * 64);  // 102400 B

    cvt_pad_kernel<<<blocks((size_t)HDIM * KPAD), 256, 0, stream>>>(Wp16, Wp, HDIM, KPAD, MD);
    cvt_pad_kernel<<<blocks((size_t)HDIM * DH), 256, 0, stream>>>(Wf16, Wf, HDIM, DH, DH + 2);
    cvt_pad_kernel<<<blocks((size_t)DEPTH * HDIM * HDIM), 256, 0, stream>>>(W1_16, W1, DEPTH * HDIM, HDIM, HDIM);
    cvt_pad_kernel<<<blocks((size_t)DEPTH * HDIM * HDIM), 256, 0, stream>>>(W2_16, W2, DEPTH * HDIM, HDIM, HDIM);
    cvt_pad_kernel<<<blocks((size_t)PDIM * HDIM), 256, 0, stream>>>(Wo16, Wo, PDIM, HDIM, HDIM);

    build_bundle_kernel<<<blocks((size_t)MROWS * KPAD), 256, 0, stream>>>(
        bundle16, U, S, Vt, friction, settle, charc, extord);

    // mode GEMM: (131072 x 96) x (96 x 512), gelu+LN -> fused16
    gemm_kernel<512, 64, EPI_LN16><<<MROWS / 64, 256, SM_512_64, stream>>>(
        bundle16, KPAD, Wp16, KPAD, bp, gp, betap,
        nullptr, nullptr, nullptr, nullptr, nullptr, fused16);

    // fuse GEMM: (8192 x 8192) x (8192 x 512) + rank2 + gelu + LN -> h32
    gemm_kernel<512, 64, EPI_FUSE><<<BN / 64, 256, SM_512_64, stream>>>(
        fused16, DH, Wf16, DH, bf, gf, betaf,
        refine, release, Wf, nullptr, h32, nullptr);

    for (int i = 0; i < DEPTH; ++i) {
        ln512_kernel<<<BN, 256, 0, stream>>>(h32, ln_g + i * HDIM, ln_b + i * HDIM, u16);
        gemm_kernel<512, 64, EPI_GELU16><<<BN / 64, 256, SM_512_64, stream>>>(
            u16, HDIM, W1_16 + (size_t)i * HDIM * HDIM, HDIM, b1 + i * HDIM,
            nullptr, nullptr, nullptr, nullptr, nullptr, nullptr, nullptr, t16);
        gemm_kernel<512, 64, EPI_RES><<<BN / 64, 256, SM_512_64, stream>>>(
            t16, HDIM, W2_16 + (size_t)i * HDIM * HDIM, HDIM, b2 + i * HDIM,
            nullptr, nullptr, nullptr, nullptr, nullptr, h32, h32, h16);
    }

    // output GEMM: (8192 x 512) x (512 x 768) + bo -> d_out
    gemm_kernel<768, 32, EPI_OUT><<<BN / 32, 256, SM_768_32, stream>>>(
        h16, HDIM, Wo16, HDIM, bo,
        nullptr, nullptr, nullptr, nullptr, nullptr, nullptr, out, nullptr);
}
